// WeightedBoxFusion_12051678232937
// MI455X (gfx1250) — matrix operand order repaired and checked by reading, not compiled, not run on hardware
//
#include <hip/hip_runtime.h>
#include <hip/hip_bf16.h>

#define N_TOT 9000
#define N_PER 3000
#define NPAD  9024          // multiple of 32 (K tiles) and 16 (M tiles)
#define W_ROW 282           // NPAD/32 bitmask words per cluster row
#define KT    282           // K tiles of 32
#define MT    564           // M tiles of 16
#define IOU_THR 0.6f

typedef __attribute__((ext_vector_type(16))) _Float16 v16h;
typedef __attribute__((ext_vector_type(8)))  float    v8f;

union HPack { unsigned int u[8]; v16h h; };
union H2    { _Float16 h[2];     unsigned int u; };

// ---------------- K1: concat boxes, apply per-model weights to scores ----------------
__global__ void k_combine(const float* __restrict__ b0, const float* __restrict__ b1,
                          const float* __restrict__ b2, const float* __restrict__ s0,
                          const float* __restrict__ s1, const float* __restrict__ s2,
                          const float* __restrict__ w,
                          float* __restrict__ cb, float* __restrict__ s) {
  int t = blockIdx.x * blockDim.x + threadIdx.x;
  if (t >= N_TOT) return;
  int m = t / N_PER, l = t % N_PER;
  const float* bp = (m == 0) ? b0 : ((m == 1) ? b1 : b2);
  const float* sp = (m == 0) ? s0 : ((m == 1) ? s1 : s2);
#pragma unroll
  for (int c = 0; c < 4; ++c) cb[t * 4 + c] = bp[l * 4 + c];
  s[t] = sp[l] * w[m];
}

// ---------------- K2: stable descending rank sort (O(N^2)), keys staged in LDS ----------------
__global__ __launch_bounds__(256) void k_sort1(const float* __restrict__ cb,
                                               const float* __restrict__ s,
                                               float* __restrict__ sb,
                                               float* __restrict__ ss) {
  __shared__ float ls[N_TOT];
  for (int t = threadIdx.x; t < N_TOT; t += 256) ls[t] = s[t];
  __syncthreads();
  int i = blockIdx.x * blockDim.x + threadIdx.x;
  if (i >= N_TOT) return;
  float si = ls[i];
  int rank = 0;
  for (int j = 0; j < N_TOT; ++j) {
    float sj = ls[j];
    rank += (sj > si) || (sj == si && j < i);
  }
  ss[rank] = si;
#pragma unroll
  for (int c = 0; c < 4; ++c) sb[rank * 4 + c] = cb[i * 4 + c];
}

// ---------------- K3: zero entire cluster bitmask (vectorized) ----------------
__global__ void k_zero(uint4* __restrict__ p, int n4) {
  int t = blockIdx.x * blockDim.x + threadIdx.x;
  if (t < n4) p[t] = make_uint4(0u, 0u, 0u, 0u);
}

// ---------------- K4: sequential clustering scan; boxes in LDS; ballot bitmask ----------------
__global__ __launch_bounds__(1024) void k_cluster(const float* __restrict__ sb,
                                                  unsigned int* __restrict__ cw) {
  __shared__ unsigned int dead[W_ROW];
  extern __shared__ float lbox[];                 // NPAD*4 floats (144 KB dynamic LDS)
  const int tid = threadIdx.x;
  const int lane = tid & 31;
  for (int t = tid; t < NPAD * 4; t += 1024) lbox[t] = sb[t];
  for (int w = tid; w < W_ROW; w += 1024) dead[w] = 0u;
  __syncthreads();

  for (int i = 0; i < N_TOT; ++i) {
    bool head = ((dead[i >> 5] >> (i & 31)) & 1u) == 0u;   // uniform across block
    if (!head) continue;   // row pre-zeroed; dead mask unchanged; no writes -> no barrier
    float x1i = lbox[i * 4 + 0], y1i = lbox[i * 4 + 1];
    float x2i = lbox[i * 4 + 2], y2i = lbox[i * 4 + 3];
    float ai = (x2i - x1i) * (y2i - y1i);
#pragma unroll
    for (int k = 0; k < 9; ++k) {
      int j = tid + k * 1024;
      int w = (tid >> 5) + k * 32;      // == j>>5 exactly
      bool member = false;
      if (j < N_TOT) {
        bool alive = ((dead[w] >> (j & 31)) & 1u) == 0u;
        if (alive) {
          if (j == i) {
            member = true;
          } else {
            float x1j = lbox[j * 4 + 0], y1j = lbox[j * 4 + 1];
            float x2j = lbox[j * 4 + 2], y2j = lbox[j * 4 + 3];
            float aj = (x2j - x1j) * (y2j - y1j);
            float xx1 = fmaxf(x1i, x1j), yy1 = fmaxf(y1i, y1j);
            float xx2 = fminf(x2i, x2j), yy2 = fminf(y2i, y2j);
            float iw = fmaxf(xx2 - xx1, 0.0f), ih = fmaxf(yy2 - yy1, 0.0f);
            float inter = iw * ih;
            // iou >= thr  <=>  inter*(1+thr) >= thr*(ai+aj)   (division-free)
            member = inter * (1.0f + IOU_THR) >= IOU_THR * (ai + aj);
          }
        }
      }
      unsigned int mask = __builtin_amdgcn_ballot_w32(member);
      if (lane == 0 && w < W_ROW) {
        cw[(size_t)i * W_ROW + w] = mask;
        unsigned int self = (w == (i >> 5)) ? (1u << (i & 31)) : 0u;
        dead[w] |= (mask & ~self);       // head stays alive; neighbors absorbed
      }
    }
    __syncthreads();
  }
}

// ---------------- K5: pack value matrix B [32xK-tile x 16] into WMMA B layout (f16) ----------------
__global__ void k_bpack(const float* __restrict__ sb, const float* __restrict__ ss,
                        unsigned int* __restrict__ bp) {
  int t = blockIdx.x * blockDim.x + threadIdx.x;
  if (t >= KT * 32) return;
  int kt = t >> 5, lane = t & 31;
  int c = lane & 15, hi = (lane >> 4) & 1;
#pragma unroll
  for (int v = 0; v < 8; ++v) {
    H2 p;
#pragma unroll
    for (int h = 0; h < 2; ++h) {
      int kk = ((v < 4) ? (hi * 8 + 2 * v) : (16 + hi * 8 + 2 * (v - 4))) + h;  // B 32x16: lane=N, v<4 -> K = hi*8 + 2v(+1); v>=4 -> K = 16 + hi*8 + 2(v-4)(+1)
      int j = kt * 32 + kk;
      float val = 0.0f;
      if (j < N_TOT) {
        if (c < 4)       val = sb[j * 4 + c] * ss[j];
        else if (c == 4) val = ss[j];
        else if (c == 5) val = 1.0f;       // counts column
      }
      p.h[h] = (_Float16)val;
    }
    bp[(size_t)t * 8 + v] = p.u;
  }
}

// ---------------- K6: cluster aggregation GEMM via v_wmma_f32_16x16x32_f16 ----------------
__global__ __launch_bounds__(256) void k_gemm(const unsigned int* __restrict__ cw,
                                              const unsigned int* __restrict__ bp,
                                              float* __restrict__ G) {
  __shared__ unsigned int lutA[4];        // 2 bits -> packed pair of f16 {0,1}
  if (threadIdx.x < 4) {
    const unsigned int v[4] = {0u, 0x00003C00u, 0x3C000000u, 0x3C003C00u};
    lutA[threadIdx.x] = v[threadIdx.x];
  }
  __syncthreads();

  int wave = threadIdx.x >> 5;
  int lane = threadIdx.x & 31;
  int tile = blockIdx.x * 8 + wave;
  if (tile >= MT) return;                 // wave-uniform guard: EXEC all-ones inside
  int m = lane & 15, hi = lane >> 4;
  int row = tile * 16 + m;
  const unsigned int* crow = cw + (size_t)row * W_ROW;
  v8f acc = {};
  for (int kt = 0; kt < KT; ++kt) {
    unsigned int aw = crow[kt];           // one bitmask word == one K-tile of 32
    HPack A;
#pragma unroll
    for (int v = 0; v < 8; ++v) {
      // A 16x32 f16 layout: v<4 -> K = hi*8 + 2v(+1); v>=4 -> K = 16 + hi*8 + 2(v-4)(+1)
      int base = (v < 4) ? (hi * 8 + 2 * v) : (16 + hi * 8 + 2 * (v - 4));
      A.u[v] = lutA[(aw >> base) & 3u];
    }
    HPack B;
    const uint4* bptr = (const uint4*)(bp + ((size_t)kt * 32 + lane) * 8);
    *(uint4*)&B.u[0] = bptr[0];
    *(uint4*)&B.u[4] = bptr[1];
    acc = __builtin_amdgcn_wmma_f32_16x16x32_f16(false, A.h, false, B.h,
                                                 (short)0, acc, false, false);
  }
#pragma unroll
  for (int d = 0; d < 8; ++d) {
    int r = tile * 16 + hi * 8 + d;       // D layout: VGPR d -> M=d (lo) / M=8+d (hi)
    G[(size_t)r * 16 + (lane & 15)] = acc[d];
  }
}

// ---------------- K7: normalize clusters, fused scores, sort keys ----------------
__global__ void k_final(const float* __restrict__ G, const float* __restrict__ w,
                        float* __restrict__ wb, float* __restrict__ wsc,
                        float* __restrict__ key) {
  int t = blockIdx.x * blockDim.x + threadIdx.x;
  if (t >= N_TOT) return;
  float wsum = w[0] + w[1] + w[2];
  float wmean = wsum * (1.0f / 3.0f);
  float sw  = G[(size_t)t * 16 + 4];
  float cnt = G[(size_t)t * 16 + 5];
  bool valid = cnt > 0.5f;
  float denom = valid ? sw : 1.0f;
#pragma unroll
  for (int c = 0; c < 4; ++c) wb[t * 4 + c] = G[(size_t)t * 16 + c] / denom;
  float sc = sw / fmaxf(wsum, wmean * cnt);
  wsc[t] = sc;
  key[t] = valid ? sc : -__builtin_inff();
}

// ---------------- K8: second stable rank sort + scatter into d_out ----------------
__global__ __launch_bounds__(256) void k_sort2(const float* __restrict__ key,
                                               const float* __restrict__ wb,
                                               const float* __restrict__ wsc,
                                               float* __restrict__ out) {
  __shared__ float lk[N_TOT];
  for (int t = threadIdx.x; t < N_TOT; t += 256) lk[t] = key[t];
  __syncthreads();
  int i = blockIdx.x * blockDim.x + threadIdx.x;
  if (i >= N_TOT) return;
  float ki = lk[i];
  int rank = 0;
  for (int j = 0; j < N_TOT; ++j) {
    float kj = lk[j];
    rank += (kj > ki) || (kj == ki && j < i);
  }
  bool valid = ki > -__builtin_inff();
  float* ob = out;                 // [9000*4] fused boxes
  float* os = out + N_TOT * 4;     // [9000] fused scores
#pragma unroll
  for (int c = 0; c < 4; ++c) ob[rank * 4 + c] = valid ? wb[i * 4 + c] : 0.0f;
  os[rank] = valid ? wsc[i] : 0.0f;
}

extern "C" void kernel_launch(void* const* d_in, const int* in_sizes, int n_in,
                              void* d_out, int out_size, void* d_ws, size_t ws_size,
                              hipStream_t stream) {
  const float* b0 = (const float*)d_in[0];
  const float* b1 = (const float*)d_in[1];
  const float* b2 = (const float*)d_in[2];
  const float* s0 = (const float*)d_in[3];
  const float* s1 = (const float*)d_in[4];
  const float* s2 = (const float*)d_in[5];
  const float* w  = (const float*)d_in[6];
  float* out = (float*)d_out;

  // workspace carve-out (256B aligned slices)
  size_t off = 0;
  auto alloc = [&](size_t bytes) -> char* {
    char* p = (char*)d_ws + off;
    off += (bytes + 255) & ~(size_t)255;
    return p;
  };
  float*        cb  = (float*)alloc((size_t)N_TOT * 4 * sizeof(float));
  float*        s   = (float*)alloc((size_t)N_TOT * sizeof(float));
  float*        sb  = (float*)alloc((size_t)NPAD * 4 * sizeof(float));
  float*        ss  = (float*)alloc((size_t)NPAD * sizeof(float));
  unsigned int* cw  = (unsigned int*)alloc((size_t)NPAD * W_ROW * sizeof(unsigned int));
  unsigned int* bp  = (unsigned int*)alloc((size_t)KT * 32 * 8 * sizeof(unsigned int));
  float*        G   = (float*)alloc((size_t)NPAD * 16 * sizeof(float));
  float*        wbx = (float*)alloc((size_t)N_TOT * 4 * sizeof(float));
  float*        wsc = (float*)alloc((size_t)N_TOT * sizeof(float));
  float*        key = (float*)alloc((size_t)N_TOT * sizeof(float));
  (void)ws_size; (void)in_sizes; (void)n_in; (void)out_size;

  const int B = 256;
  const int gN = (N_TOT + B - 1) / B;                         // 36
  const int n4 = (NPAD * W_ROW) / 4;                          // 636192 uint4
  const int gZero = (n4 + B - 1) / B;
  const int gBp = (KT * 32 + B - 1) / B;                      // 36
  const int gGemm = (MT + 7) / 8;                             // 71 blocks x 8 waves
  const size_t ldsBox = (size_t)NPAD * 4 * sizeof(float);     // 144 KB dynamic LDS

  k_combine<<<gN, B, 0, stream>>>(b0, b1, b2, s0, s1, s2, w, cb, s);
  k_sort1  <<<gN, B, 0, stream>>>(cb, s, sb, ss);
  k_zero   <<<gZero, B, 0, stream>>>((uint4*)cw, n4);
  k_cluster<<<1, 1024, ldsBox, stream>>>(sb, cw);
  k_bpack  <<<gBp, B, 0, stream>>>(sb, ss, bp);
  k_gemm   <<<gGemm, B, 0, stream>>>(cw, bp, G);
  k_final  <<<gN, B, 0, stream>>>(G, w, wbx, wsc, key);
  k_sort2  <<<gN, B, 0, stream>>>(key, wbx, wsc, out);
}
